// GATv2ConvModel_51831665328521
// MI455X (gfx1250) — matrix operand order repaired and checked by reading, compiled, not run on hardware
//
#include <hip/hip_runtime.h>
#include <cmath>

typedef __attribute__((ext_vector_type(16))) _Float16 v16h;
typedef __attribute__((ext_vector_type(8)))  _Float16 v8h;
typedef __attribute__((ext_vector_type(8)))  float    v8f;
typedef __attribute__((ext_vector_type(4)))  float    f32x4;

#define NEG_SLOPE 0.2f

__device__ __forceinline__ void atomic_max_float(float* addr, float v) {
  // sign-punned float max: correct for mixed signs when initialized to -inf
  if (v >= 0.0f) atomicMax((int*)addr, __float_as_int(v));
  else           atomicMin((unsigned int*)addr, __float_as_uint(v));
}

// ---------------------------------------------------------------------------
// Prep: WT[n*K + k] = (f16) W[k*Nc + n]   (W is K x Nc, Nc == 64; tiny)
// Makes B fragments two contiguous 8xf16 runs -> vector loads, no in-loop cvt.
// ---------------------------------------------------------------------------
__global__ void __launch_bounds__(256)
transpose_w_f16(const float* __restrict__ W, _Float16* __restrict__ WT,
                int K, int Nc) {
  const int t = blockIdx.x * blockDim.x + threadIdx.x;
  if (t >= K * Nc) return;
  const int k = t / Nc, n = t % Nc;
  WT[n * K + k] = (_Float16)W[t];
}

// ---------------------------------------------------------------------------
// out[M,64] = H[M,K] @ W[K,64] + bias ; W given as f16 WT[64][K].
// One wave -> 16 rows x 64 cols (4 accumulators, A fragment reused 4x).
// 128-thread block -> 64 rows. K % 32 == 0, M % 16 == 0.
// A frag (16x32 f16): lane {lmod=M row, half=K half}; elem i ->
//   K = (i/8)*16 + half*8 + (i%8)  => two contiguous float4-pairs from H.
// B frag (32x16 f16): N=lane%16; elem i ->
//   K = kb + (i/8)*16 + half*8 + (i%8) => two contiguous 8xf16 runs from WT row.
// C/D: reg r, lane l -> M = r + 8*(l/16), N = l%16.
// ---------------------------------------------------------------------------
__global__ void __launch_bounds__(128)
gemm_bias_wmma(const float* __restrict__ H, const _Float16* __restrict__ WT,
               const float* __restrict__ bias, float* __restrict__ out,
               int M, int K) {
  constexpr int Nc = 64;
  const int wave = threadIdx.x >> 5;
  const int lane = threadIdx.x & 31;
  const int m0 = (blockIdx.x * 4 + wave) << 4;
  if (m0 >= M) return;
  const int half = lane >> 4;
  const int lmod = lane & 15;

  v8f acc[4] = {};
  const float* hrow = H + (size_t)(m0 + lmod) * K;

  for (int kb = 0; kb < K; kb += 32) {
    // ---- A fragment: two contiguous 8-float runs -> 4x global_load_b128
    const f32x4 a0 = *(const f32x4*)(hrow + kb + half * 8);
    const f32x4 a1 = *(const f32x4*)(hrow + kb + half * 8 + 4);
    const f32x4 a2 = *(const f32x4*)(hrow + kb + 16 + half * 8);
    const f32x4 a3 = *(const f32x4*)(hrow + kb + 16 + half * 8 + 4);
    v16h a;
#pragma unroll
    for (int i = 0; i < 4; ++i) {
      a[i]      = (_Float16)a0[i];
      a[4 + i]  = (_Float16)a1[i];
      a[8 + i]  = (_Float16)a2[i];
      a[12 + i] = (_Float16)a3[i];
    }
    // ---- 4 N-tiles share the A fragment
#pragma unroll
    for (int nt = 0; nt < 4; ++nt) {
      const _Float16* wp = WT + (size_t)(nt * 16 + lmod) * K + kb + half * 8;
      const v8h b0 = *(const v8h*)(wp);
      const v8h b1 = *(const v8h*)(wp + 16);
      v16h b;
#pragma unroll
      for (int i = 0; i < 8; ++i) { b[i] = b0[i]; b[8 + i] = b1[i]; }
      acc[nt] = __builtin_amdgcn_wmma_f32_16x16x32_f16(
          false, a, false, b, (short)0, acc[nt], false, false);
    }
  }
#pragma unroll
  for (int nt = 0; nt < 4; ++nt) {
#pragma unroll
    for (int r = 0; r < 8; ++r) {
      const int mm = m0 + r + (half << 3);
      const int nn = nt * 16 + lmod;
      out[(size_t)mm * Nc + nn] = acc[nt][r] + bias[nn];
    }
  }
}

// out <- bias (per column), m <- -inf, s <- 0
__global__ void __launch_bounds__(256)
init_node(float* __restrict__ out, const float* __restrict__ bias,
          float* __restrict__ mval, float* __restrict__ sval, int N_) {
  constexpr int C = 64;
  const int t = blockIdx.x * blockDim.x + threadIdx.x;
  if (t < N_ * C) out[t] = bias[t & (C - 1)];
  if (t < N_) { mval[t] = -INFINITY; sval[t] = 0.0f; }
}

// Pass 1: one wave per edge; lane owns channels {2l, 2l+1} (one b64 per gather)
__global__ void __launch_bounds__(256)
edge_logits(const int* __restrict__ ei, const float* __restrict__ ea,
            const float* __restrict__ xl, const float* __restrict__ xr,
            const float* __restrict__ We, const float* __restrict__ att,
            float* __restrict__ logit, float* __restrict__ mval,
            int E_, int N_) {
  constexpr int C = 64;
  const int wid  = (int)((blockIdx.x * blockDim.x + threadIdx.x) >> 5);
  const int lane = threadIdx.x & 31;
  const int Et = E_ + N_;
  if (wid >= Et) return;
  int s, d; float eav;
  if (wid < E_) { s = ei[wid]; d = ei[E_ + wid]; eav = ea[wid]; }
  else          { s = wid - E_; d = s; eav = 1.0f; }   // self-loop, attr fill 1
  const int c = lane << 1;
  const float2 vl = *(const float2*)(xl + (size_t)s * C + c);
  const float2 vr = *(const float2*)(xr + (size_t)d * C + c);
  const float2 we = *(const float2*)(We + c);
  const float2 at = *(const float2*)(att + c);
  float z0 = vl.x + vr.x + eav * we.x;
  float z1 = vl.y + vr.y + eav * we.y;
  z0 = (z0 > 0.0f) ? z0 : z0 * NEG_SLOPE;
  z1 = (z1 > 0.0f) ? z1 : z1 * NEG_SLOPE;
  float acc = fmaf(z0, at.x, z1 * at.y);
#pragma unroll
  for (int off = 16; off > 0; off >>= 1) acc += __shfl_down(acc, off, 32);
  if (lane == 0) {
    logit[wid] = acc;
    atomic_max_float(&mval[d], acc);
  }
}

// Pass 2: one thread per edge. ex = exp(logit - m[dst]); s[dst] += ex
__global__ void __launch_bounds__(256)
edge_exp(const int* __restrict__ ei, const float* __restrict__ logit,
         const float* __restrict__ mval, float* __restrict__ exv,
         float* __restrict__ sval, int E_, int N_) {
  const int t = blockIdx.x * blockDim.x + threadIdx.x;
  const int Et = E_ + N_;
  if (t >= Et) return;
  const int d = (t < E_) ? ei[E_ + t] : (t - E_);
  const float ex = __expf(logit[t] - mval[d]);
  exv[t] = ex;
  atomicAdd(&sval[d], ex);
}

// Pass 3: one wave per edge. out[dst] += (ex/s[dst]) * xl[src]
__global__ void __launch_bounds__(256)
edge_scatter(const int* __restrict__ ei, const float* __restrict__ xl,
             const float* __restrict__ exv, const float* __restrict__ sval,
             float* __restrict__ out, int E_, int N_) {
  constexpr int C = 64;
  const int wid  = (int)((blockIdx.x * blockDim.x + threadIdx.x) >> 5);
  const int lane = threadIdx.x & 31;
  const int Et = E_ + N_;
  if (wid >= Et) return;
  int s, d;
  if (wid < E_) { s = ei[wid]; d = ei[E_ + wid]; }
  else          { s = wid - E_; d = s; }
  const float alpha = exv[wid] / sval[d];
  const int c = lane << 1;
  const float2 vl = *(const float2*)(xl + (size_t)s * C + c);
  atomicAdd(&out[(size_t)d * C + c],     alpha * vl.x);
  atomicAdd(&out[(size_t)d * C + c + 1], alpha * vl.y);
}

extern "C" void kernel_launch(void* const* d_in, const int* in_sizes, int n_in,
                              void* d_out, int out_size, void* d_ws, size_t ws_size,
                              hipStream_t stream) {
  const float* x    = (const float*)d_in[0];
  const int*   ei   = (const int*)d_in[1];
  const float* ea   = (const float*)d_in[2];
  const float* Wl0  = (const float*)d_in[3];
  const float* bl0  = (const float*)d_in[4];
  const float* Wr0  = (const float*)d_in[5];
  const float* br0  = (const float*)d_in[6];
  const float* We0  = (const float*)d_in[7];
  const float* att0 = (const float*)d_in[8];
  const float* bi0  = (const float*)d_in[9];
  const float* Wl1  = (const float*)d_in[10];
  const float* bl1  = (const float*)d_in[11];
  const float* Wr1  = (const float*)d_in[12];
  const float* br1  = (const float*)d_in[13];
  const float* We1  = (const float*)d_in[14];
  const float* att1 = (const float*)d_in[15];
  const float* bi1  = (const float*)d_in[16];

  const int CIN0 = 128, C = 64;
  const int N_ = in_sizes[0] / CIN0;      // 50000
  const int E_ = in_sizes[2];             // 1,000,000 (edge_attr is E x 1)
  const int Et = E_ + N_;

  // workspace carve (~47 MB floats + 32 KB f16): fits in the 192MB L2
  float* p   = (float*)d_ws;
  float* xl  = p; p += (size_t)N_ * C;
  float* xr  = p; p += (size_t)N_ * C;
  float* h1  = p; p += (size_t)N_ * C;
  float* lg  = p; p += (size_t)Et;
  float* exv = p; p += (size_t)Et;
  float* mv  = p; p += (size_t)N_;
  float* sv  = p; p += (size_t)N_;
  _Float16* wtl = (_Float16*)p;           // 64*128 f16
  _Float16* wtr = wtl + (size_t)C * CIN0; // 64*128 f16

  const int gemmBlocks = (N_ / 16 + 3) / 4;     // 4 wave-tiles (64 rows)/block
  const int nodeBlocks = (N_ * C + 255) / 256;
  const int waveBlocks = (Et + 7) / 8;          // 8 waves per 256-thread block
  const int edgeBlocks = (Et + 255) / 256;

  float* out = (float*)d_out;

  // ---- layer 0 (K = 128) ----
  transpose_w_f16<<<(CIN0 * C + 255) / 256, 256, 0, stream>>>(Wl0, wtl, CIN0, C);
  transpose_w_f16<<<(CIN0 * C + 255) / 256, 256, 0, stream>>>(Wr0, wtr, CIN0, C);
  gemm_bias_wmma<<<gemmBlocks, 128, 0, stream>>>(x, wtl, bl0, xl, N_, CIN0);
  gemm_bias_wmma<<<gemmBlocks, 128, 0, stream>>>(x, wtr, br0, xr, N_, CIN0);
  init_node<<<nodeBlocks, 256, 0, stream>>>(h1, bi0, mv, sv, N_);
  edge_logits<<<waveBlocks, 256, 0, stream>>>(ei, ea, xl, xr, We0, att0, lg, mv, E_, N_);
  edge_exp<<<edgeBlocks, 256, 0, stream>>>(ei, lg, mv, exv, sv, E_, N_);
  edge_scatter<<<waveBlocks, 256, 0, stream>>>(ei, xl, exv, sv, h1, E_, N_);

  // ---- layer 1 (K = 64) ----
  transpose_w_f16<<<(C * C + 255) / 256, 256, 0, stream>>>(Wl1, wtl, C, C);
  transpose_w_f16<<<(C * C + 255) / 256, 256, 0, stream>>>(Wr1, wtr, C, C);
  gemm_bias_wmma<<<gemmBlocks, 128, 0, stream>>>(h1, wtl, bl1, xl, N_, C);
  gemm_bias_wmma<<<gemmBlocks, 128, 0, stream>>>(h1, wtr, br1, xr, N_, C);
  init_node<<<nodeBlocks, 256, 0, stream>>>(out, bi1, mv, sv, N_);
  edge_logits<<<waveBlocks, 256, 0, stream>>>(ei, ea, xl, xr, We1, att1, lg, mv, E_, N_);
  edge_exp<<<edgeBlocks, 256, 0, stream>>>(ei, lg, mv, exv, sv, E_, N_);
  edge_scatter<<<waveBlocks, 256, 0, stream>>>(ei, xl, exv, sv, out, E_, N_);
}
